// RecurrentModule_1030792151619
// MI455X (gfx1250) — matrix operand order repaired and checked by reading, not compiled, not run on hardware
//
#include <hip/hip_runtime.h>
#include <hip/hip_bf16.h>

typedef __bf16 bf16;
typedef __attribute__((ext_vector_type(16))) __bf16 v16bf;
typedef __attribute__((ext_vector_type(8)))  __bf16 v8bf;
typedef __attribute__((ext_vector_type(8)))  float  v8f;

#define B_  64
#define S_  512
#define D_  256
#define H_  512
#define G_  (3 * H_)   // 1536

#define GRU_BLOCKS   48                    // 384 waves == 384 gh tiles (4 x 96)
#define GRU_THREADS  (GRU_BLOCKS * 256)    // 12288

// ---------------------------------------------------------------------------
// Fragment loaders (bf16 WMMA operands; this code's contiguous k order, same for A and B)
//   A 16x32: lane l -> row l%16, K-chunk  [(l/16)*16,+16) contiguous
//   B 32x16: lane l -> col l%16, K-chunk  [(l/16)*16,+16) contiguous (B stored N x K)
// ---------------------------------------------------------------------------
__device__ __forceinline__ v16bf load_frag_a(const bf16* __restrict__ row, int k0, int lh)
{
    return *(const v16bf*)(row + k0 + lh * 16);
}

__device__ __forceinline__ v16bf load_frag_b(const bf16* __restrict__ row, int k0, int lh)
{
    return *(const v16bf*)(row + k0 + lh * 16);
}

// ---------------------------------------------------------------------------
// 4x4 register-blocked GEMM: one wave computes a 64x64 block of C.
// A [M,K] bf16 row-major; Bt [N,K] bf16 row-major; C f32 row-major.
// __launch_bounds__(256, 1): allocate registers for 1 wave-group/EU so the
// 16 accumulator tiles (128 VGPRs) stay register-resident (no scratch spill).
// ---------------------------------------------------------------------------
__global__ void __launch_bounds__(256, 1) k_gemm_blk(
    const bf16* __restrict__ A, int lda, long long strideA,
    const bf16* __restrict__ Bt, int ldb, long long strideB,
    float* __restrict__ C, int ldc, long long strideC,
    int Nb /* 64-wide blocks along N */, int K,
    const float* __restrict__ bias, int blocksPerBatch)
{
    const int wave  = blockIdx.x * (blockDim.x >> 5) + (threadIdx.x >> 5);
    const int lane  = threadIdx.x & 31;
    const int batch = wave / blocksPerBatch;
    const int t     = wave % blocksPerBatch;
    const int m0 = (t / Nb) * 64;
    const int n0 = (t % Nb) * 64;
    const int lr = lane & 15, lh = lane >> 4;

    const bf16* __restrict__ Ab = A  + (long long)batch * strideA;
    const bf16* __restrict__ Bb = Bt + (long long)batch * strideB;
    float* __restrict__       Cb = C  + (long long)batch * strideC;

    const bf16* arow[4];
    const bf16* brow[4];
#pragma unroll
    for (int i = 0; i < 4; ++i) {
        arow[i] = Ab + (long long)(m0 + i * 16 + lr) * lda;
        brow[i] = Bb + (long long)(n0 + i * 16 + lr) * ldb;
    }

    v8f c[4][4] = {};
    for (int k0 = 0; k0 < K; k0 += 32) {
        v16bf b[4];
#pragma unroll
        for (int j = 0; j < 4; ++j) b[j] = load_frag_b(brow[j], k0, lh);
#pragma unroll
        for (int i = 0; i < 4; ++i) {
            v16bf a = load_frag_a(arow[i], k0, lh);
#pragma unroll
            for (int j = 0; j < 4; ++j)
                c[i][j] = __builtin_amdgcn_wmma_f32_16x16x32_bf16(
                    false, a, false, b[j], (short)0, c[i][j], false, false);
        }
    }

#pragma unroll
    for (int j = 0; j < 4; ++j) {
        const int col = n0 + j * 16 + lr;
        const float bv = bias ? bias[col] : 0.0f;
#pragma unroll
        for (int i = 0; i < 4; ++i)
#pragma unroll
            for (int r = 0; r < 8; ++r)
                Cb[(long long)(m0 + i * 16 + r + lh * 8) * ldc + col] = c[i][j][r] + bv;
    }
}

// ---------------------------------------------------------------------------
// Grid barrier (L2 atomics; bar[0]=arrive count, bar[1]=generation)
// ---------------------------------------------------------------------------
__device__ __forceinline__ void grid_barrier(unsigned* __restrict__ bar)
{
    __syncthreads();
    if (threadIdx.x == 0) {
        __threadfence();
        unsigned g = __hip_atomic_load(&bar[1], __ATOMIC_RELAXED, __HIP_MEMORY_SCOPE_AGENT);
        unsigned a = __hip_atomic_fetch_add(&bar[0], 1u, __ATOMIC_ACQ_REL, __HIP_MEMORY_SCOPE_AGENT);
        if (a == (unsigned)(gridDim.x - 1)) {
            __hip_atomic_store(&bar[0], 0u, __ATOMIC_RELAXED, __HIP_MEMORY_SCOPE_AGENT);
            __hip_atomic_fetch_add(&bar[1], 1u, __ATOMIC_RELEASE, __HIP_MEMORY_SCOPE_AGENT);
        } else {
            while (__hip_atomic_load(&bar[1], __ATOMIC_ACQUIRE, __HIP_MEMORY_SCOPE_AGENT) == g)
                __builtin_amdgcn_s_sleep(2);
        }
        __threadfence();
    }
    __syncthreads();
}

// ---------------------------------------------------------------------------
// Persistent GRU: all 512 steps in one launch.
// Each wave owns one 16x16 tile of gh = h_prev @ W_hh^T + b_hh and keeps its
// 16 W_hh B-fragments resident in VGPRs for the whole sequence.
// ---------------------------------------------------------------------------
__global__ void __launch_bounds__(256, 1) k_gru_persistent(
    const float* __restrict__ xp,     // [B*S, 3H]  (b_ih already added)
    const bf16*  __restrict__ Whh_b,  // [3H, H] bf16 (Bt layout)
    const float* __restrict__ b_hh,   // [3H]
    float* __restrict__ gh,           // [B, 3H] scratch
    float* __restrict__ hpf,          // [B, H] f32 state
    bf16*  __restrict__ hpb,          // [B, H] bf16 state
    bf16*  __restrict__ hall,         // [B, S, H]
    bf16*  __restrict__ hT,           // [B, H, S]
    unsigned* __restrict__ bar)
{
    const int wave = blockIdx.x * 8 + (threadIdx.x >> 5);  // 0..383
    const int lane = threadIdx.x & 31;
    const int mi = wave / (G_ / 16);                       // 0..3
    const int ni = wave % (G_ / 16);                       // 0..95
    const int m0 = mi * 16, n0 = ni * 16;
    const int lr = lane & 15, lh = lane >> 4;
    const int gtid = blockIdx.x * 256 + threadIdx.x;       // 0..12287

    // Preload the 16 invariant W_hh fragments for this tile (stay in VGPRs).
    const bf16* brow = Whh_b + (long long)(n0 + lr) * H_;
    v16bf bfr[H_ / 32];
#pragma unroll
    for (int kk = 0; kk < H_ / 32; ++kk) bfr[kk] = load_frag_b(brow, kk * 32, lh);

    const bf16* arow = hpb + (long long)(m0 + lr) * H_;
    const float bv = b_hh[n0 + lr];
    float* gout = gh + (long long)(m0 + lh * 8) * G_ + n0 + lr;

    for (int s = 0; s < S_; ++s) {
        // ---- GEMM phase: gh tile = h_prev @ W_hh^T + b_hh ----
        v8f c = {};
#pragma unroll
        for (int kk = 0; kk < H_ / 32; ++kk) {
            v16bf a = load_frag_a(arow, kk * 32, lh);
            c = __builtin_amdgcn_wmma_f32_16x16x32_bf16(
                    false, a, false, bfr[kk], (short)0, c, false, false);
        }
#pragma unroll
        for (int r = 0; r < 8; ++r) gout[(long long)r * G_] = c[r] + bv;

        grid_barrier(bar);

        // ---- gate phase ----
        for (int idx = gtid; idx < B_ * H_; idx += GRU_THREADS) {
            const int b = idx >> 9;            // / H_
            const int j = idx & (H_ - 1);
            const float* xr = xp + (long long)(b * S_ + s) * G_;
            const float* gr = gh + (long long)b * G_;
            float r = 1.0f / (1.0f + __expf(-(xr[j] + gr[j])));
            float z = 1.0f / (1.0f + __expf(-(xr[H_ + j] + gr[H_ + j])));
            float n = tanhf(xr[2 * H_ + j] + r * gr[2 * H_ + j]);
            float h = (1.0f - z) * n + z * hpf[idx];
            hpf[idx] = h;
            bf16 hb = (bf16)h;
            hpb[idx] = hb;
            hall[(long long)(b * S_ + s) * H_ + j] = hb;
            hT[((long long)b * H_ + j) * S_ + s] = hb;
        }

        grid_barrier(bar);
    }
}

// ---------------------------------------------------------------------------
// Elementwise helpers
// ---------------------------------------------------------------------------
__global__ void k_f32_to_bf16(const float* __restrict__ x, bf16* __restrict__ y, int n)
{
    int i = blockIdx.x * blockDim.x + threadIdx.x;
    if (i < n) y[i] = (bf16)x[i];
}

__global__ void k_init_state(float* __restrict__ hf, bf16* __restrict__ hb,
                             unsigned* __restrict__ bar, int n)
{
    int i = blockIdx.x * blockDim.x + threadIdx.x;
    if (i < n) { hf[i] = 0.0f; hb[i] = (bf16)0.0f; }
    if (i < 2) bar[i] = 0u;
}

// Row softmax over full row, then tril mask (unrenormalized), -> bf16.
__global__ void k_softmax_mask(const float* __restrict__ scores,
                               bf16* __restrict__ alphas)
{
    const int row = blockIdx.x;          // b*S + s
    const int s   = row & (S_ - 1);
    const float* x = scores + (long long)row * S_;
    bf16* y = alphas + (long long)row * S_;

    __shared__ float red_max[8];
    __shared__ float red_sum[8];
    const int lane = threadIdx.x & 31;
    const int wv   = threadIdx.x >> 5;

    float m = -3.4e38f;
    for (int t = threadIdx.x; t < S_; t += 256) m = fmaxf(m, x[t]);
#pragma unroll
    for (int off = 16; off > 0; off >>= 1) m = fmaxf(m, __shfl_xor(m, off, 32));
    if (lane == 0) red_max[wv] = m;
    __syncthreads();
    m = red_max[0];
#pragma unroll
    for (int i = 1; i < 8; ++i) m = fmaxf(m, red_max[i]);

    float sum = 0.0f;
    for (int t = threadIdx.x; t < S_; t += 256) sum += __expf(x[t] - m);
#pragma unroll
    for (int off = 16; off > 0; off >>= 1) sum += __shfl_xor(sum, off, 32);
    if (lane == 0) red_sum[wv] = sum;
    __syncthreads();
    sum = 0.0f;
#pragma unroll
    for (int i = 0; i < 8; ++i) sum += red_sum[i];
    const float inv = 1.0f / sum;

    for (int t = threadIdx.x; t < S_; t += 256) {
        float a = __expf(x[t] - m) * inv;
        y[t] = (t <= s) ? (bf16)a : (bf16)0.0f;
    }
}

// ---------------------------------------------------------------------------
// Host side
// ---------------------------------------------------------------------------
extern "C" void kernel_launch(void* const* d_in, const int* in_sizes, int n_in,
                              void* d_out, int out_size, void* d_ws, size_t ws_size,
                              hipStream_t stream)
{
    const float* inp  = (const float*)d_in[0];   // [B,S,D]
    // d_in[1]: seq_lens (all == S; reference reduces to reshape) - unused
    const float* W_ih = (const float*)d_in[2];   // [3H, D]
    const float* W_hh = (const float*)d_in[3];   // [3H, H]
    const float* b_ih = (const float*)d_in[4];   // [3H]
    const float* b_hh = (const float*)d_in[5];   // [3H]
    float* out = (float*)d_out;                  // [B*S, H]

    char* ws = (char*)d_ws;
    size_t off = 0;
    bf16* inp_b  = (bf16*)(ws + off); off += (size_t)B_ * S_ * D_ * 2;       // 16 MB
    bf16* Wih_b  = (bf16*)(ws + off); off += (size_t)G_ * D_ * 2;            // 768 KB
    bf16* Whh_b  = (bf16*)(ws + off); off += (size_t)G_ * H_ * 2;            // 1.5 MB
    float* xp    = (float*)(ws + off); off += (size_t)B_ * S_ * G_ * 4;      // 192 MB
    float* gh    = (float*)(ws + off); off += (size_t)B_ * G_ * 4;           // 384 KB
    float* hpf   = (float*)(ws + off); off += (size_t)B_ * H_ * 4;           // 128 KB
    bf16* hpb    = (bf16*)(ws + off); off += (size_t)B_ * H_ * 2;            // 64 KB
    bf16* hall   = (bf16*)(ws + off); off += (size_t)B_ * S_ * H_ * 2;       // 32 MB
    bf16* hT     = (bf16*)(ws + off); off += (size_t)B_ * H_ * S_ * 2;       // 32 MB
    float* sc    = (float*)(ws + off); off += (size_t)B_ * S_ * S_ * 4;      // 64 MB
    bf16* alph   = (bf16*)(ws + off); off += (size_t)B_ * S_ * S_ * 2;       // 32 MB
    unsigned* bar = (unsigned*)(ws + off); off += 256;
    (void)ws_size; (void)in_sizes; (void)n_in; (void)out_size;

    // 1) fp32 -> bf16 conversions + state/barrier init
    {
        int n = B_ * S_ * D_;
        k_f32_to_bf16<<<(n + 255) / 256, 256, 0, stream>>>(inp, inp_b, n);
        n = G_ * D_;
        k_f32_to_bf16<<<(n + 255) / 256, 256, 0, stream>>>(W_ih, Wih_b, n);
        n = G_ * H_;
        k_f32_to_bf16<<<(n + 255) / 256, 256, 0, stream>>>(W_hh, Whh_b, n);
        n = B_ * H_;
        k_init_state<<<(n + 255) / 256, 256, 0, stream>>>(hpf, hpb, bar, n);
    }

    // 2) x_proj = inp @ W_ih^T + b_ih   [32768 x 1536], K=256  (4x4 blocked)
    {
        const int Mb = (B_ * S_) / 64, Nb = G_ / 64;   // 512 x 24
        const int blocks = Mb * Nb;                    // 12288 waves
        k_gemm_blk<<<blocks / 8, 256, 0, stream>>>(
            inp_b, D_, 0, Wih_b, D_, 0, xp, G_, 0, Nb, D_, b_ih, blocks);
    }

    // 3) GRU recurrence: one persistent launch, 512 steps
    k_gru_persistent<<<GRU_BLOCKS, 256, 0, stream>>>(
        xp, Whh_b, b_hh, gh, hpf, hpb, hall, hT, bar);

    // 4) gram = h @ h^T per batch  [64 x (512x512)], K=512  (4x4 blocked)
    {
        const int Nb = S_ / 64;                       // 8
        const int perBatch = (S_ / 64) * Nb;          // 64
        const int waves = B_ * perBatch;              // 4096
        k_gemm_blk<<<waves / 8, 256, 0, stream>>>(
            hall, H_, (long long)S_ * H_,
            hall, H_, (long long)S_ * H_,
            sc,   S_, (long long)S_ * S_,
            Nb, H_, nullptr, perBatch);
    }

    // 5) full-row softmax + tril mask -> bf16 alphas
    k_softmax_mask<<<B_ * S_, 256, 0, stream>>>(sc, alph);

    // 6) out = alphas @ h per batch (B operand is h^T, maintained by GRU)
    {
        const int Nb = H_ / 64;                       // 8
        const int perBatch = (S_ / 64) * Nb;          // 64
        const int waves = B_ * perBatch;              // 4096
        k_gemm_blk<<<waves / 8, 256, 0, stream>>>(
            alph, S_, (long long)S_ * S_,
            hT,   S_, (long long)H_ * S_,
            out,  H_, (long long)S_ * H_,
            Nb, S_, nullptr, perBatch);
    }
}
